// MANSF_37469294690809
// MI455X (gfx1250) — matrix operand order repaired and checked by reading, not compiled, not run on hardware
//
#include <hip/hip_runtime.h>
#include <hip/hip_bf16.h>

// MANSF pipeline for MI455X (gfx1250, wave32, WMMA).
// Sizes: T=5, N=2048, K=20, E=512, H=64, BL=64, G1=G2=64, U=8.
// lin_attn == sum over time axis (softmax over singleton axis == 1.0), so the
// ap_/am_/as_ parameters are dead and each attention is a plain reduction.

typedef __attribute__((ext_vector_type(16))) _Float16 v16h;
typedef __attribute__((ext_vector_type(8)))  float    v8f;

union Frag { v16h v; uint4 q[2]; };

static __device__ __forceinline__ float sigm(float x) { return 1.f / (1.f + __expf(-x)); }

// ---------------------------------------------------------------------------
// Generic f16-WMMA GEMM: C[u] = A[u] * B[u]^T(+bias)
//   block = 256 threads (8 waves), tile 128(M) x 64(N), K step 32.
//   B_NK=true : B stored [N][K] row-major (e.g. torch Linear weight)
//   B_NK=false: B stored [K][N] row-major (transposed into LDS)
// A is fp32 in memory, converted to f16 while staging (f32 accumulate).
// ---------------------------------------------------------------------------
template<bool B_NK>
__global__ __launch_bounds__(256) void gemm_wmma(
    const float* __restrict__ A, int lda, long long strideAu,
    const float* __restrict__ B, int ldb, long long strideBu,
    const float* __restrict__ bias,
    float* __restrict__ C, int ldc, long long strideCu,
    int Kdim)
{
  constexpr int P = 40;                 // f16 pitch: 32 data + 8 pad
  __shared__ _Float16 As[128 * P];
  __shared__ _Float16 Bs[64 * P];

  const int m0 = blockIdx.x * 128;
  const int n0 = blockIdx.y * 64;
  const int u  = blockIdx.z;
  A += (long long)u * strideAu;
  B += (long long)u * strideBu;
  C += (long long)u * strideCu;

  const int tid  = threadIdx.x;
  const int w    = tid >> 5;            // wave id = M-tile (0..7)
  const int lane = tid & 31;
  const int lm   = lane & 15;
  const int hf   = lane >> 4;

  v8f acc[4] = {};

  for (int kc = 0; kc < Kdim; kc += 32) {
    { // stage A tile 128x32 (row-major, f32 -> f16)
      const int r  = tid >> 1;
      const int k0 = (tid & 1) * 16;
      const float* src = A + (long long)(m0 + r) * lda + kc + k0;
      _Float16* dst = &As[r * P + k0];
      #pragma unroll
      for (int v = 0; v < 4; ++v) {
        float4 f = reinterpret_cast<const float4*>(src)[v];
        dst[4*v+0] = (_Float16)f.x; dst[4*v+1] = (_Float16)f.y;
        dst[4*v+2] = (_Float16)f.z; dst[4*v+3] = (_Float16)f.w;
      }
    }
    if (B_NK) { // B[n][k] -> Bs[n][k]
      const int nn = tid >> 2;
      const int k0 = (tid & 3) * 8;
      const float* src = B + (long long)(n0 + nn) * ldb + kc + k0;
      _Float16* dst = &Bs[nn * P + k0];
      #pragma unroll
      for (int v = 0; v < 2; ++v) {
        float4 f = reinterpret_cast<const float4*>(src)[v];
        dst[4*v+0] = (_Float16)f.x; dst[4*v+1] = (_Float16)f.y;
        dst[4*v+2] = (_Float16)f.z; dst[4*v+3] = (_Float16)f.w;
      }
    } else {    // B[k][n] -> Bs[n][k]  (transpose on LDS write)
      const int kr  = tid >> 3;
      const int nn0 = (tid & 7) * 8;
      const float* src = B + (long long)(kc + kr) * ldb + n0 + nn0;
      #pragma unroll
      for (int v = 0; v < 2; ++v) {
        float4 f = reinterpret_cast<const float4*>(src)[v];
        Bs[(nn0 + 4*v + 0) * P + kr] = (_Float16)f.x;
        Bs[(nn0 + 4*v + 1) * P + kr] = (_Float16)f.y;
        Bs[(nn0 + 4*v + 2) * P + kr] = (_Float16)f.z;
        Bs[(nn0 + 4*v + 3) * P + kr] = (_Float16)f.w;
      }
    }
    __syncthreads();
    // A fragment (16x32 f16): elems 0..7 = K hf*8.., elems 8..15 = K 16+hf*8..
    Frag a;
    a.q[0] = *reinterpret_cast<const uint4*>(&As[(w*16 + lm) * P + hf*8]);
    a.q[1] = *reinterpret_cast<const uint4*>(&As[(w*16 + lm) * P + 16 + hf*8]);
    #pragma unroll
    for (int nt = 0; nt < 4; ++nt) {
      // B fragment (32x16): lane col = lm, elems 0..7 = K hf*8.., elems 8..15 = K 16+hf*8..
      Frag b;
      b.q[0] = *reinterpret_cast<const uint4*>(&Bs[(nt*16 + lm) * P + hf*8]);
      b.q[1] = *reinterpret_cast<const uint4*>(&Bs[(nt*16 + lm) * P + 16 + hf*8]);
      acc[nt] = __builtin_amdgcn_wmma_f32_16x16x32_f16(
          false, a.v, false, b.v, (short)0, acc[nt], false, false);
    }
    __syncthreads();
  }
  #pragma unroll
  for (int nt = 0; nt < 4; ++nt) {
    const int col = n0 + nt*16 + lm;
    const float bv = bias ? bias[col] : 0.f;
    #pragma unroll
    for (int e = 0; e < 8; ++e) {
      const int row = m0 + w*16 + hf*8 + e;   // C/D layout: M = 8*hf + e
      C[(long long)row * ldc + col] = acc[nt][e] + bv;
    }
  }
}

// ---------------------------------------------------------------------------
// Small GRU (full T=5 sequence in one kernel) + time-sum output.
// block = 8 rows x 64 threads; thread j owns h[j] of its row.
// Whh staged in LDS (stride 65 -> conflict-free); Wih read via cache.
// ---------------------------------------------------------------------------
template<int IN>
__global__ __launch_bounds__(512) void gru_seq(
    const float* __restrict__ x, int stride_n, int stride_t,
    const float* __restrict__ Wih, const float* __restrict__ Whh,
    const float* __restrict__ bih, const float* __restrict__ bhh,
    float* __restrict__ sumout)
{
  __shared__ float WhhL[192 * 65];
  __shared__ float bihL[192], bhhL[192];
  __shared__ float h[8][64];
  __shared__ float xs[8][IN];

  const int tid = threadIdx.x;
  const int lr  = tid >> 6;
  const int j   = tid & 63;
  const int n0  = blockIdx.x * 8;

  for (int idx = tid; idx < 192 * 64; idx += 512)
    WhhL[(idx >> 6) * 65 + (idx & 63)] = Whh[idx];
  if (tid < 192) { bihL[tid] = bih[tid]; bhhL[tid] = bhh[tid]; }
  h[lr][j] = 0.f;
  float sacc = 0.f;
  __syncthreads();

  for (int t = 0; t < 5; ++t) {
    for (int idx = tid; idx < 8 * IN; idx += 512) {
      int rr = idx / IN, ii = idx % IN;
      xs[rr][ii] = x[(long long)(n0 + rr) * stride_n + t * stride_t + ii];
    }
    __syncthreads();
    float gir = bihL[j], giz = bihL[64 + j], gin = bihL[128 + j];
    #pragma unroll 4
    for (int i = 0; i < IN; ++i) {
      float xi = xs[lr][i];
      gir += xi * Wih[j * IN + i];
      giz += xi * Wih[(64 + j) * IN + i];
      gin += xi * Wih[(128 + j) * IN + i];
    }
    float ghr = bhhL[j], ghz = bhhL[64 + j], ghn = bhhL[128 + j];
    #pragma unroll 4
    for (int i = 0; i < 64; ++i) {
      float hi = h[lr][i];
      ghr += hi * WhhL[j * 65 + i];
      ghz += hi * WhhL[(64 + j) * 65 + i];
      ghn += hi * WhhL[(128 + j) * 65 + i];
    }
    float rg = sigm(gir + ghr);
    float zg = sigm(giz + ghz);
    float ng = tanhf(gin + rg * ghn);
    float hn = (1.f - zg) * ng + zg * h[lr][j];
    __syncthreads();
    h[lr][j] = hn;
    sacc += hn;
    __syncthreads();
  }
  sumout[(long long)(n0 + lr) * 64 + j] = sacc;
}

// ---------------------------------------------------------------------------
// Recurrent part of gru_m for one t-slice: gi (with bih) precomputed by WMMA
// GEMM; 20 steps with h kept in LDS. Writes r[n][t][:] = sum_k h_k.
// ---------------------------------------------------------------------------
__global__ __launch_bounds__(512) void gru_m_rec(
    const float* __restrict__ gi, const float* __restrict__ Whh,
    const float* __restrict__ bhh, float* __restrict__ rout, int t)
{
  __shared__ float WhhL[192 * 65];
  __shared__ float bhhL[192];
  __shared__ float h[8][64];

  const int tid = threadIdx.x;
  const int lr  = tid >> 6;
  const int j   = tid & 63;
  const int n   = blockIdx.x * 8 + lr;

  for (int idx = tid; idx < 192 * 64; idx += 512)
    WhhL[(idx >> 6) * 65 + (idx & 63)] = Whh[idx];
  if (tid < 192) bhhL[tid] = bhh[tid];
  h[lr][j] = 0.f;
  float racc = 0.f;
  __syncthreads();

  for (int k = 0; k < 20; ++k) {
    const float* gp = gi + ((long long)n * 20 + k) * 192;
    float gir = gp[j], giz = gp[64 + j], gin = gp[128 + j];
    float ghr = bhhL[j], ghz = bhhL[64 + j], ghn = bhhL[128 + j];
    #pragma unroll 4
    for (int i = 0; i < 64; ++i) {
      float hi = h[lr][i];
      ghr += hi * WhhL[j * 65 + i];
      ghz += hi * WhhL[(64 + j) * 65 + i];
      ghn += hi * WhhL[(128 + j) * 65 + i];
    }
    float rg = sigm(gir + ghr);
    float zg = sigm(giz + ghz);
    float ng = tanhf(gin + rg * ghn);
    float hn = (1.f - zg) * ng + zg * h[lr][j];
    __syncthreads();
    h[lr][j] = hn;
    racc += hn;
    __syncthreads();
  }
  rout[((long long)n * 5 + t) * 64 + j] = racc;
}

// x[n,o] = relu(blend_b[o] + sum_i q[n,i] * tmp[n, o*64+i])
__global__ __launch_bounds__(256) void blend_reduce(
    const float* __restrict__ tmp, const float* __restrict__ q,
    const float* __restrict__ bb, float* __restrict__ xout)
{
  int idx = blockIdx.x * 256 + threadIdx.x;   // 2048*64 threads
  int n = idx >> 6, o = idx & 63;
  const float* qn = q + (long long)n * 64;
  const float* tp = tmp + (long long)n * 4096 + o * 64;
  float acc = bb[o];
  #pragma unroll 4
  for (int i = 0; i < 64; ++i) acc += qn[i] * tp[i];
  xout[idx] = fmaxf(acc, 0.f);
}

// s1[u,n] = Wh[u,n,:] . a[u,0:64];  s2[u,n] = Wh[u,n,:] . a[u,64:128]
__global__ __launch_bounds__(256) void s12_kernel(
    const float* __restrict__ Wh, const float* __restrict__ a,
    float* __restrict__ s1, float* __restrict__ s2)
{
  int idx = blockIdx.x * 256 + threadIdx.x;   // 8*2048
  int u = idx >> 11, n = idx & 2047;
  const float* wp = Wh + ((long long)u * 2048 + n) * 64;
  const float* a1 = a + u * 128;
  const float* a2 = a1 + 64;
  float r1 = 0.f, r2 = 0.f;
  #pragma unroll 4
  for (int o = 0; o < 64; ++o) { float w = wp[o]; r1 += w * a1[o]; r2 += w * a2[o]; }
  s1[idx] = r1; s2[idx] = r2;
}

// Per-row online softmax stats for all 8 heads at once (adj read once).
// block = 8 waves, one wave per row i; masked entries contribute exactly 0.
__global__ __launch_bounds__(256) void rowstats(
    const float* __restrict__ adj, const float* __restrict__ s1,
    const float* __restrict__ s2, float* __restrict__ rmax,
    float* __restrict__ rden)
{
  const int w    = threadIdx.x >> 5;
  const int lane = threadIdx.x & 31;
  const int i    = blockIdx.x * 8 + w;

  float s1v[8], m[8], s[8];
  #pragma unroll
  for (int u = 0; u < 8; ++u) { s1v[u] = s1[u * 2048 + i]; m[u] = -3.402823e38f; s[u] = 0.f; }

  for (int j = lane; j < 2048; j += 32) {
    float a = adj[(long long)i * 2048 + j];
    if (a > 0.f) {
      #pragma unroll
      for (int u = 0; u < 8; ++u) {
        float v = s1v[u] + s2[u * 2048 + j];
        v = v > 0.f ? v : 0.01f * v;                 // leaky_relu slope 0.01
        if (v > m[u]) { s[u] = s[u] * __expf(m[u] - v) + 1.f; m[u] = v; }
        else          { s[u] += __expf(v - m[u]); }
      }
    }
  }
  for (int off = 16; off; off >>= 1) {
    #pragma unroll
    for (int u = 0; u < 8; ++u) {
      float m2 = __shfl_xor(m[u], off);
      float t2 = __shfl_xor(s[u], off);
      float M  = fmaxf(m[u], m2);
      s[u] = s[u] * __expf(m[u] - M) + t2 * __expf(m2 - M);
      m[u] = M;
    }
  }
  if (lane == 0) {
    #pragma unroll
    for (int u = 0; u < 8; ++u) { rmax[u * 2048 + i] = m[u]; rden[u * 2048 + i] = s[u]; }
  }
}

// ---------------------------------------------------------------------------
// SGAT attention-weighted aggregation via WMMA (flash style):
// out[n, u*64+o] = act( (1/den_i) * sum_j P[i,j] * Wh[u,j,o] )
// P tile (32x32, f16) built in LDS each chunk; Wh chunk transposed into LDS.
// ACT: 0 = ELU(alpha=1), 1 = sigmoid.
// grid = (2048/32, U), block = 256 (8 waves -> 2 Mtiles x 4 Ntiles).
// ---------------------------------------------------------------------------
template<int ACT>
__global__ __launch_bounds__(256) void sgat_pw(
    const float* __restrict__ adj, const float* __restrict__ s1,
    const float* __restrict__ s2, const float* __restrict__ rmax,
    const float* __restrict__ rden, const float* __restrict__ Wh,
    float* __restrict__ out)
{
  constexpr int P = 40;
  __shared__ _Float16 Ps[32 * P];
  __shared__ _Float16 Ws[64 * P];

  const int i0 = blockIdx.x * 32;
  const int u  = blockIdx.y;
  const float* s1u = s1 + u * 2048;
  const float* s2u = s2 + u * 2048;
  const float* mxu = rmax + u * 2048;
  const float* dnu = rden + u * 2048;
  const float* Whu = Wh + (long long)u * 2048 * 64;

  const int tid  = threadIdx.x;
  const int w    = tid >> 5;
  const int lane = tid & 31;
  const int lm   = lane & 15;
  const int hf   = lane >> 4;
  const int mt   = w >> 2;     // 0..1
  const int nt   = w & 3;      // 0..3

  v8f acc = {};

  for (int jb = 0; jb < 2048; jb += 32) {
    { // stage P: exp(lrelu(s1_i + s2_j) - rowmax_i) where adj>0 else 0
      const int r  = tid >> 3;
      const int j0 = (tid & 7) * 4;
      float4 av  = *reinterpret_cast<const float4*>(&adj[(long long)(i0 + r) * 2048 + jb + j0]);
      float4 sv  = *reinterpret_cast<const float4*>(&s2u[jb + j0]);
      float s1v  = s1u[i0 + r];
      float mrow = mxu[i0 + r];
      float pv[4]; float aa[4] = {av.x, av.y, av.z, av.w};
      float ss[4] = {sv.x, sv.y, sv.z, sv.w};
      #pragma unroll
      for (int q2 = 0; q2 < 4; ++q2) {
        float e = s1v + ss[q2];
        e = e > 0.f ? e : 0.01f * e;
        pv[q2] = (aa[q2] > 0.f) ? __expf(e - mrow) : 0.f;
      }
      _Float16* dst = &Ps[r * P + j0];
      dst[0] = (_Float16)pv[0]; dst[1] = (_Float16)pv[1];
      dst[2] = (_Float16)pv[2]; dst[3] = (_Float16)pv[3];
    }
    { // stage Wh chunk transposed: Ws[o][j]
      const int jr = tid >> 3;
      const int o0 = (tid & 7) * 8;
      const float* src = Whu + (long long)(jb + jr) * 64 + o0;
      float4 f0 = reinterpret_cast<const float4*>(src)[0];
      float4 f1 = reinterpret_cast<const float4*>(src)[1];
      Ws[(o0 + 0) * P + jr] = (_Float16)f0.x; Ws[(o0 + 1) * P + jr] = (_Float16)f0.y;
      Ws[(o0 + 2) * P + jr] = (_Float16)f0.z; Ws[(o0 + 3) * P + jr] = (_Float16)f0.w;
      Ws[(o0 + 4) * P + jr] = (_Float16)f1.x; Ws[(o0 + 5) * P + jr] = (_Float16)f1.y;
      Ws[(o0 + 6) * P + jr] = (_Float16)f1.z; Ws[(o0 + 7) * P + jr] = (_Float16)f1.w;
    }
    __syncthreads();
    Frag a, b;
    a.q[0] = *reinterpret_cast<const uint4*>(&Ps[(mt*16 + lm) * P + hf*8]);
    a.q[1] = *reinterpret_cast<const uint4*>(&Ps[(mt*16 + lm) * P + 16 + hf*8]);
    b.q[0] = *reinterpret_cast<const uint4*>(&Ws[(nt*16 + lm) * P + hf*8]);
    b.q[1] = *reinterpret_cast<const uint4*>(&Ws[(nt*16 + lm) * P + 16 + hf*8]);
    acc = __builtin_amdgcn_wmma_f32_16x16x32_f16(
        false, a.v, false, b.v, (short)0, acc, false, false);
    __syncthreads();
  }
  #pragma unroll
  for (int e = 0; e < 8; ++e) {
    const int row = i0 + mt*16 + hf*8 + e;
    float v = acc[e] / dnu[row];
    if (ACT == 0) v = v > 0.f ? v : (__expf(v) - 1.f);     // ELU(1.0)
    else          v = sigm(v);                              // sigmoid
    out[(long long)row * 512 + u * 64 + nt*16 + lm] = v;
  }
}

// out[n] = sigmoid(fin_b + nz[n,:].fin_W)
__global__ __launch_bounds__(256) void final_kernel(
    const float* __restrict__ nz, const float* __restrict__ finW,
    const float* __restrict__ finb, float* __restrict__ out)
{
  __shared__ float wsh[512];
  const int tid = threadIdx.x;
  for (int i = tid; i < 512; i += 256) wsh[i] = finW[i];
  __syncthreads();
  const int n = blockIdx.x * 256 + tid;
  const float* zp = nz + (long long)n * 512;
  float acc = finb[0];
  #pragma unroll 4
  for (int i = 0; i < 512; ++i) acc += zp[i] * wsh[i];
  out[n] = sigm(acc);
}

// ---------------------------------------------------------------------------
extern "C" void kernel_launch(void* const* d_in, const int* in_sizes, int n_in,
                              void* d_out, int out_size, void* d_ws, size_t ws_size,
                              hipStream_t stream) {
  (void)in_sizes; (void)n_in; (void)out_size; (void)ws_size;
  const float* p     = (const float*)d_in[0];
  const float* m     = (const float*)d_in[1];
  const float* adj   = (const float*)d_in[2];
  const float* Wih_p = (const float*)d_in[3];
  const float* Whh_p = (const float*)d_in[4];
  const float* bih_p = (const float*)d_in[5];
  const float* bhh_p = (const float*)d_in[6];
  const float* Wih_m = (const float*)d_in[7];
  const float* Whh_m = (const float*)d_in[8];
  const float* bih_m = (const float*)d_in[9];
  const float* bhh_m = (const float*)d_in[10];
  const float* Wih_s = (const float*)d_in[11];
  const float* Whh_s = (const float*)d_in[12];
  const float* bih_s = (const float*)d_in[13];
  const float* bhh_s = (const float*)d_in[14];
  // d_in[15..23]: lin_attn params — mathematically dead (softmax over size-1 axis)
  const float* blend_W = (const float*)d_in[24];
  const float* blend_b = (const float*)d_in[25];
  const float* g1_W  = (const float*)d_in[26];
  const float* g1_a  = (const float*)d_in[27];
  const float* g2_W  = (const float*)d_in[28];
  const float* g2_a  = (const float*)d_in[29];
  const float* fin_W = (const float*)d_in[30];
  const float* fin_b = (const float*)d_in[31];
  float* out = (float*)d_out;

  // workspace layout (floats); gi and tmp share one region (disjoint in time)
  float* wsf  = (float*)d_ws;
  float* gi   = wsf;                       // 40960*192 = 7,864,320
  float* tmp  = wsf;                       // 2048*4096 = 8,388,608 (max)
  size_t o = 8388608;
  float* q    = wsf + o; o += 131072;      // [2048][64]
  float* rbuf = wsf + o; o += 655360;      // [2048][5][64]
  float* c    = wsf + o; o += 131072;      // [2048][64]
  float* xbl  = wsf + o; o += 131072;      // [2048][64]
  float* Wh   = wsf + o; o += 1048576;     // [8][2048][64] (reused layer1/2)
  float* s1b  = wsf + o; o += 16384;
  float* s2b  = wsf + o; o += 16384;
  float* rmx  = wsf + o; o += 16384;
  float* rdn  = wsf + o; o += 16384;
  float* zb   = wsf + o; o += 1048576;     // [2048][512]
  float* nzb  = wsf + o; o += 1048576;     // [2048][512]

  // 1) GRU over p -> q = sum_t h
  gru_seq<3><<<256, 512, 0, stream>>>(p, 15, 3, Wih_p, Whh_p, bih_p, bhh_p, q);

  // 2) gru_m per t-slice: gi GEMM (WMMA, 40 GFLOP total) + recurrent kernel
  for (int t = 0; t < 5; ++t) {
    gemm_wmma<true><<<dim3(320, 3, 1), 256, 0, stream>>>(
        m + (long long)t * 2048 * 20 * 512, 512, 0,
        Wih_m, 512, 0, bih_m, gi, 192, 0, 512);
    gru_m_rec<<<256, 512, 0, stream>>>(gi, Whh_m, bhh_m, rbuf, t);
  }

  // 3) GRU over r -> c = sum_t h
  gru_seq<64><<<256, 512, 0, stream>>>(rbuf, 320, 64, Wih_s, Whh_s, bih_s, bhh_s, c);

  // 4) bilinear blend: tmp[n,(o,i)] = sum_j blend_W[o,i,j] c[n,j]  (WMMA)
  gemm_wmma<true><<<dim3(16, 64, 1), 256, 0, stream>>>(
      c, 64, 0, blend_W, 64, 0, nullptr, tmp, 4096, 0, 64);
  blend_reduce<<<512, 256, 0, stream>>>(tmp, q, blend_b, xbl);

  // 5) SGAT layer 1
  gemm_wmma<false><<<dim3(16, 1, 8), 256, 0, stream>>>(
      xbl, 64, 0, g1_W, 64, 64 * 64, nullptr, Wh, 64, 2048 * 64, 64);
  s12_kernel<<<64, 256, 0, stream>>>(Wh, g1_a, s1b, s2b);
  rowstats<<<256, 256, 0, stream>>>(adj, s1b, s2b, rmx, rdn);
  sgat_pw<0><<<dim3(64, 8), 256, 0, stream>>>(adj, s1b, s2b, rmx, rdn, Wh, zb);

  // 6) SGAT layer 2
  gemm_wmma<false><<<dim3(16, 1, 8), 256, 0, stream>>>(
      zb, 512, 0, g2_W, 64, 512 * 64, nullptr, Wh, 64, 2048 * 64, 512);
  s12_kernel<<<64, 256, 0, stream>>>(Wh, g2_a, s1b, s2b);
  rowstats<<<256, 256, 0, stream>>>(adj, s1b, s2b, rmx, rdn);
  sgat_pw<1><<<dim3(64, 8), 256, 0, stream>>>(adj, s1b, s2b, rmx, rdn, Wh, nzb);

  // 7) final projection + sigmoid
  final_kernel<<<8, 256, 0, stream>>>(nzb, fin_W, fin_b, out);
}
